// SingleHeadTransformer_86191403696359
// MI455X (gfx1250) — compile-verified, matrix operand order checked by reading, not run on hardware
//
#include <hip/hip_runtime.h>

// ---------------------------------------------------------------------------
// Single-head causal transformer block + LM head for MI455X (gfx1250).
// All GEMMs run through one WMMA (v_wmma_f32_16x16x32_bf16) NT kernel:
//   C[M x N] = A[M x K] * B[N x K]^T   (both operands row-major over K)
// v3b: global->LDS staging uses GLOBAL_LOAD_ASYNC_TO_LDS_B128 (ASYNCcnt) so
//     no staging VGPRs are held across the WMMA block (v2 spilled to scratch).
//     Builtin prototype (from hipcc diagnostic) takes int4-vector pointers:
//     (v4i AS1*, v4i AS3*, imm offset, imm cpol). Wave tile 64x64 (16 WMMA /
//     K-step), WG tile 128x256, LDS double buffer, one s_wait_asynccnt +
//     barrier per K-step, global_prefetch for tile t+2. bf16 Wo^T (98.5 MB)
//     stays L2-resident.
// ---------------------------------------------------------------------------

typedef __bf16 bf16;
typedef __bf16 v8bf  __attribute__((ext_vector_type(8)));
typedef __bf16 v16bf __attribute__((ext_vector_type(16)));
typedef float  v8f   __attribute__((ext_vector_type(8)));
typedef int    v4i_  __attribute__((vector_size(16)));   // matches builtin proto

#define B_    2
#define T_    2048
#define C_    1024
#define H_    1024
#define V_    50257
#define VPAD_ 50432   // V rounded up to multiple of 256 (GEMM BN)

#define HAVE_ASYNC_LDS __has_builtin(__builtin_amdgcn_global_load_async_to_lds_b128)

#define GAS1(p) ((__attribute__((address_space(1))) v4i_*)(p))
#define LAS3(p) ((__attribute__((address_space(3))) v4i_*)(p))

__device__ __forceinline__ void wait_async0() {
#if __has_builtin(__builtin_amdgcn_s_wait_asynccnt)
  __builtin_amdgcn_s_wait_asynccnt(0);
#else
  asm volatile("s_wait_asynccnt 0x0" ::: "memory");
#endif
}

__device__ __forceinline__ bf16 f2bf(float f) {
  // round-to-nearest-even fp32 -> bf16
  unsigned u = __builtin_bit_cast(unsigned, f);
  u += 0x7FFFu + ((u >> 16) & 1u);
  unsigned short h = (unsigned short)(u >> 16);
  return __builtin_bit_cast(bf16, h);
}

// ---------------- embeddings: x = tok_emb[idx] + pos_emb, to bf16 ----------
__global__ void embed_kernel(const int* __restrict__ idx,
                             const float* __restrict__ tok,
                             const float* __restrict__ pos,
                             bf16* __restrict__ xbf) {
  size_t i = (size_t)blockIdx.x * blockDim.x + threadIdx.x;  // < B*T*C
  int c  = (int)(i & (C_ - 1));
  int bt = (int)(i >> 10);          // C_ == 1024
  int t  = bt & (T_ - 1);
  int tokid = idx[bt];
  float v = tok[(size_t)tokid * C_ + c] + pos[(size_t)t * C_ + c];
  xbf[i] = f2bf(v);
}

// ------------- fp32 [K x N] -> bf16 [Npad x K] transpose+convert -----------
__global__ void convT_f32_bf16(const float* __restrict__ in,
                               bf16* __restrict__ out,
                               int K, int N) {
  __shared__ float tile[32][33];
  int n0 = blockIdx.x * 32;
  int k0 = blockIdx.y * 32;
  int tx = threadIdx.x & 31;
  int ty = threadIdx.x >> 5;        // 0..7
#pragma unroll
  for (int s = 0; s < 32; s += 8) {
    int k = k0 + ty + s, n = n0 + tx;
    tile[ty + s][tx] = (n < N) ? in[(size_t)k * N + n] : 0.f;   // pad cols -> 0
  }
  __syncthreads();
#pragma unroll
  for (int s = 0; s < 32; s += 8) {
    int n = n0 + ty + s, k = k0 + tx;
    out[(size_t)n * K + k] = f2bf(tile[tx][ty + s]);
  }
}

// ---------------- bf16 [R x C] -> [C x R] batched transpose ----------------
__global__ void transpose_bf16(const bf16* __restrict__ in,
                               bf16* __restrict__ out,
                               int R, int Ccols) {
  __shared__ bf16 tile[32][33];
  size_t bOff = (size_t)blockIdx.z * R * Ccols;
  in += bOff;
  out += bOff;
  int c0 = blockIdx.x * 32, r0 = blockIdx.y * 32;
  int tx = threadIdx.x & 31, ty = threadIdx.x >> 5;
#pragma unroll
  for (int s = 0; s < 32; s += 8)
    tile[ty + s][tx] = in[(size_t)(r0 + ty + s) * Ccols + c0 + tx];
  __syncthreads();
#pragma unroll
  for (int s = 0; s < 32; s += 8)
    out[(size_t)(c0 + ty + s) * R + r0 + tx] = tile[tx][ty + s];
}

// ------------- causal softmax over fp32 scores, write bf16 P ---------------
__global__ void softmax_causal(const float* __restrict__ S, bf16* __restrict__ P) {
  int t = blockIdx.x, b = blockIdx.y;
  const float* row = S + ((size_t)b * T_ + t) * T_;
  bf16* prow = P + ((size_t)b * T_ + t) * T_;
  int len = t + 1;
  __shared__ float red[256];
  int tid = threadIdx.x;
  float m = -3.4e38f;
  for (int s = tid; s < len; s += 256) m = fmaxf(m, row[s]);
  red[tid] = m;
  __syncthreads();
  for (int o = 128; o > 0; o >>= 1) {
    if (tid < o) red[tid] = fmaxf(red[tid], red[tid + o]);
    __syncthreads();
  }
  m = red[0];
  __syncthreads();
  float sum = 0.f;
  for (int s = tid; s < len; s += 256) sum += __expf(row[s] - m);
  red[tid] = sum;
  __syncthreads();
  for (int o = 128; o > 0; o >>= 1) {
    if (tid < o) red[tid] += red[tid + o];
    __syncthreads();
  }
  float inv = 1.0f / red[0];
  for (int s = tid; s < T_; s += 256) {
    float p = (s < len) ? __expf(row[s] - m) * inv : 0.f;   // zero above diagonal
    prow[s] = f2bf(p);
  }
}

// ---------------------------------------------------------------------------
// NT WMMA GEMM: C[M x N] = alpha * A[M x K] * B[N x K]^T (+ bias)
//  - 256 threads = 8 wave32 arranged 2 (M) x 4 (N); wave tile 64x64
//    => 16 v_wmma_f32_16x16x32_bf16 per K-step per wave
//  - WG tile BM=128 x BN=256, K-step 32, LDS row stride 40 elems (80B) to
//    keep the per-lane ds_load_b128 fragment reads bank-conflict free
//  - double-buffered LDS filled by async global->LDS DMA (no staging VGPRs,
//    no spills); drain with s_wait_asynccnt 0 + barrier once per K-step
//  - global_prefetch (builtin) pulls tile t+2 toward L2
//  - causalSkip: drop tiles fully above the diagonal (scores GEMM)
//  - causalK:    bound the K loop at the diagonal (attn @ v; P is 0 above it)
// ---------------------------------------------------------------------------
template <bool OUT_BF16, bool HAS_BIAS>
__global__ __launch_bounds__(256)
void gemm_nt_wmma(const bf16* __restrict__ A, const bf16* __restrict__ Bm,
                  void* __restrict__ Cout, const float* __restrict__ bias,
                  int M, int Nstore, int K,
                  int lda, int ldb, int ldc,
                  long long aBatch, long long bBatch, long long cBatch,
                  float alpha, int causalSkip, int causalK) {
  constexpr int BM = 128, BN = 256, BK = 32, LDW = 40;
  __shared__ alignas(16) bf16 As[2][BM * LDW];
  __shared__ alignas(16) bf16 Bs[2][BN * LDW];

  const int tid  = threadIdx.x;
  const int lane = tid & 31;
  const int wave = tid >> 5;   // 0..7
  const int wr   = wave >> 2;  // 0..1  (64 rows each)
  const int wc   = wave & 3;   // 0..3  (64 cols each)
  const int mBlock = blockIdx.y * BM;
  const int nBlock = blockIdx.x * BN;

  if (causalSkip && nBlock > mBlock + (BM - 1)) return;  // fully masked tile

  const size_t zb = blockIdx.z;
  A  += zb * (size_t)aBatch;
  Bm += zb * (size_t)bBatch;

  int kEnd = K;
  if (causalK) { int lim = mBlock + BM; kEnd = lim < K ? lim : K; }

  // global->LDS staging: chunk c covers row (c>>2), 8-elem col block (c&3)
  const int rowS = tid >> 2;       // 0..63, rows at rowS + 64*i
  const int colS = (tid & 3) * 8;  // 0,8,16,24 (bf16 elems; 16B chunks)

  // Async DMA tile copy: hardware writes LDS directly (ASYNCcnt), so no
  // VGPRs are occupied across the WMMA block.
  auto copyTile = [&](int buf, int k0) {
#if HAVE_ASYNC_LDS
#pragma unroll
    for (int i = 0; i < 2; ++i)
      __builtin_amdgcn_global_load_async_to_lds_b128(
          GAS1(A + (size_t)(mBlock + rowS + 64 * i) * lda + k0 + colS),
          LAS3(&As[buf][(rowS + 64 * i) * LDW + colS]),
          0, 0);
#pragma unroll
    for (int i = 0; i < 4; ++i)
      __builtin_amdgcn_global_load_async_to_lds_b128(
          GAS1(Bm + (size_t)(nBlock + rowS + 64 * i) * ldb + k0 + colS),
          LAS3(&Bs[buf][(rowS + 64 * i) * LDW + colS]),
          0, 0);
#else
    // fallback: synchronous VGPR-staged copy (short live range, no spill)
#pragma unroll
    for (int i = 0; i < 2; ++i)
      *(uint4*)(&As[buf][(rowS + 64 * i) * LDW + colS]) =
          *(const uint4*)(A + (size_t)(mBlock + rowS + 64 * i) * lda + k0 + colS);
#pragma unroll
    for (int i = 0; i < 4; ++i)
      *(uint4*)(&Bs[buf][(rowS + 64 * i) * LDW + colS]) =
          *(const uint4*)(Bm + (size_t)(nBlock + rowS + 64 * i) * ldb + k0 + colS);
#endif
  };

  v8f acc[4][4];
#pragma unroll
  for (int i = 0; i < 4; ++i)
#pragma unroll
    for (int j = 0; j < 4; ++j)
      acc[i][j] = (v8f){0.f, 0.f, 0.f, 0.f, 0.f, 0.f, 0.f, 0.f};

  // CDNA5 16-bit A/B fragment layout: lanes 0-15 take K 0..7 & 16..23,
  // lanes 16-31 take K 8..15 & 24..31; row/col = lane & 15.
  const int kOff = (lane < 16) ? 0 : 8;
  const int rsel = lane & 15;

  const int nTiles = kEnd / BK;  // kEnd is always a multiple of BK here

  copyTile(0, 0);
  wait_async0();
  __syncthreads();

  for (int t = 0; t < nTiles; ++t) {
    const int cur = t & 1;

    if (t + 1 < nTiles) {
      copyTile(cur ^ 1, (t + 1) * BK);   // DMA overlaps this step's WMMAs
      if (t + 2 < nTiles) {              // keep L2 ahead of the double buffer
        __builtin_prefetch(A + (size_t)(mBlock + rowS) * lda + (t + 2) * BK + colS, 0, 1);
        __builtin_prefetch(Bm + (size_t)(nBlock + rowS) * ldb + (t + 2) * BK + colS, 0, 1);
      }
    }

    v16bf afrag[4], bfrag[4];
#pragma unroll
    for (int i = 0; i < 4; ++i) {
      const bf16* p = &As[cur][(wr * 64 + i * 16 + rsel) * LDW + kOff];
      v8bf lo = *(const v8bf*)p;
      v8bf hi = *(const v8bf*)(p + 16);
      afrag[i] = __builtin_shufflevector(lo, hi, 0, 1, 2, 3, 4, 5, 6, 7, 8, 9,
                                         10, 11, 12, 13, 14, 15);
    }
#pragma unroll
    for (int j = 0; j < 4; ++j) {
      const bf16* p = &Bs[cur][(wc * 64 + j * 16 + rsel) * LDW + kOff];
      v8bf lo = *(const v8bf*)p;
      v8bf hi = *(const v8bf*)(p + 16);
      bfrag[j] = __builtin_shufflevector(lo, hi, 0, 1, 2, 3, 4, 5, 6, 7, 8, 9,
                                         10, 11, 12, 13, 14, 15);
    }
#pragma unroll
    for (int i = 0; i < 4; ++i)
#pragma unroll
      for (int j = 0; j < 4; ++j)
        acc[i][j] = __builtin_amdgcn_wmma_f32_16x16x32_bf16(
            false, afrag[i], false, bfrag[j], (short)0, acc[i][j], false, false);

    // all our reads of buf[cur] are consumed by the WMMAs above (dscnt),
    // so after draining our DMAs and the barrier, buf[cur^1] is ready and
    // buf[cur] may be overwritten next step.
    wait_async0();
    __syncthreads();
  }

  // store: C/D layout -> VGPR e holds row (e + 8*(lane>=16)), col = lane&15
  const int mhalf = (lane >= 16) ? 8 : 0;
  const size_t cbase = zb * (size_t)cBatch;
#pragma unroll
  for (int i = 0; i < 4; ++i) {
#pragma unroll
    for (int j = 0; j < 4; ++j) {
      int col = nBlock + wc * 64 + j * 16 + rsel;
      if (col >= Nstore) continue;           // V = 50257 tail guard
      float bv = HAS_BIAS ? bias[col] : 0.f;
#pragma unroll
      for (int e = 0; e < 8; ++e) {
        int row = mBlock + wr * 64 + i * 16 + mhalf + e;
        float val = acc[i][j][e] * alpha + bv;
        if (OUT_BF16)
          ((bf16*)Cout)[cbase + (size_t)row * ldc + col] = f2bf(val);
        else
          ((float*)Cout)[cbase + (size_t)row * ldc + col] = val;
      }
    }
  }
}

// ---------------------------------------------------------------------------
extern "C" void kernel_launch(void* const* d_in, const int* in_sizes, int n_in,
                              void* d_out, int out_size, void* d_ws,
                              size_t ws_size, hipStream_t stream) {
  const int*   idx = (const int*)d_in[0];
  const float* tok = (const float*)d_in[1];
  const float* pos = (const float*)d_in[2];
  const float* Wk  = (const float*)d_in[3];
  const float* Wq  = (const float*)d_in[4];
  const float* Wv  = (const float*)d_in[5];
  const float* Wo  = (const float*)d_in[6];
  const float* bo  = (const float*)d_in[7];
  float* out = (float*)d_out;

  char* ws = (char*)d_ws;
  size_t off = 0;
  auto alloc = [&](size_t bytes) {
    size_t r = off;
    off = (off + bytes + 255) & ~(size_t)255;
    return r;
  };
  const size_t BT = (size_t)B_ * T_;  // 4096

  bf16*  xbf = (bf16*)(ws + alloc(BT * C_ * 2));
  bf16*  q   = (bf16*)(ws + alloc(BT * H_ * 2));
  bf16*  kk  = (bf16*)(ws + alloc(BT * H_ * 2));
  bf16*  v   = (bf16*)(ws + alloc(BT * H_ * 2));
  bf16*  vT  = (bf16*)(ws + alloc(BT * H_ * 2));
  bf16*  ao  = (bf16*)(ws + alloc(BT * H_ * 2));
  bf16*  WqT = (bf16*)(ws + alloc((size_t)C_ * H_ * 2));
  bf16*  WkT = (bf16*)(ws + alloc((size_t)C_ * H_ * 2));
  bf16*  WvT = (bf16*)(ws + alloc((size_t)C_ * H_ * 2));
  bf16*  WoT = (bf16*)(ws + alloc((size_t)VPAD_ * H_ * 2));  // L2-resident
  float* sc  = (float*)(ws + alloc((size_t)B_ * T_ * T_ * 4));
  bf16*  P   = (bf16*)(ws + alloc((size_t)B_ * T_ * T_ * 2));

  // 1) embeddings -> bf16 x
  embed_kernel<<<dim3((unsigned)(BT * C_ / 256)), 256, 0, stream>>>(idx, tok, pos, xbf);

  // 2) weights: fp32 [K x N] -> bf16 [Npad x K]
  convT_f32_bf16<<<dim3(H_ / 32, C_ / 32), 256, 0, stream>>>(Wq, WqT, C_, H_);
  convT_f32_bf16<<<dim3(H_ / 32, C_ / 32), 256, 0, stream>>>(Wk, WkT, C_, H_);
  convT_f32_bf16<<<dim3(H_ / 32, C_ / 32), 256, 0, stream>>>(Wv, WvT, C_, H_);
  convT_f32_bf16<<<dim3(VPAD_ / 32, H_ / 32), 256, 0, stream>>>(Wo, WoT, H_, V_);

  // 3) q = (x Wq) / sqrt(H)  (scale folded), k = x Wk, v = x Wv
  dim3 gq(H_ / 256, (unsigned)(BT / 128), 1);
  gemm_nt_wmma<true, false><<<gq, 256, 0, stream>>>(
      xbf, WqT, q, nullptr, (int)BT, H_, C_, C_, C_, H_, 0, 0, 0, 0.03125f, 0, 0);
  gemm_nt_wmma<true, false><<<gq, 256, 0, stream>>>(
      xbf, WkT, kk, nullptr, (int)BT, H_, C_, C_, C_, H_, 0, 0, 0, 1.f, 0, 0);
  gemm_nt_wmma<true, false><<<gq, 256, 0, stream>>>(
      xbf, WvT, v, nullptr, (int)BT, H_, C_, C_, C_, H_, 0, 0, 0, 1.f, 0, 0);

  // 4) v^T per batch (for NT attn@v)
  transpose_bf16<<<dim3(H_ / 32, T_ / 32, B_), 256, 0, stream>>>(v, vT, T_, H_);

  // 5) scores = q k^T (tiles fully above the causal diagonal skipped)
  gemm_nt_wmma<false, false><<<dim3(T_ / 256, T_ / 128, B_), 256, 0, stream>>>(
      q, kk, sc, nullptr, T_, T_, H_, H_, H_, T_,
      (long long)T_ * H_, (long long)T_ * H_, (long long)T_ * T_, 1.f, 1, 0);

  // 6) causal softmax -> bf16 P (zeros above diagonal)
  softmax_causal<<<dim3(T_, B_), 256, 0, stream>>>(sc, P);

  // 7) attn_out = P v  (K loop bounded at the diagonal)
  gemm_nt_wmma<true, false><<<dim3(H_ / 256, T_ / 128, B_), 256, 0, stream>>>(
      P, vT, ao, nullptr, T_, H_, T_, T_, T_, H_,
      (long long)T_ * T_, (long long)H_ * T_, (long long)T_ * H_, 1.f, 0, 1);

  // 8) logits = attn_out Wo + bo  (fp32 out, Wo^T bf16 served from L2)
  gemm_nt_wmma<false, true><<<dim3(VPAD_ / 256, (unsigned)(BT / 128), 1), 256, 0, stream>>>(
      ao, WoT, out, bo, (int)BT, V_, C_, C_, C_, V_, 0, 0, 0, 1.f, 0, 0);
}
